// MultiHeadLatentAttention_4123168604840
// MI455X (gfx1250) — matrix operand order repaired and checked by reading, compiled, not run on hardware
//
#include <hip/hip_runtime.h>

// ---------------------------------------------------------------------------
// MLA forward for MI455X (gfx1250, wave32, WMMA bf16 16x16x32, f32 accum)
//   B=1, S=2048, HIDDEN=2048, NUM_HEADS=16, MLA_DIM=HEAD_DIM=128
// Uses: v_wmma_f32_16x16x32_bf16, global_load_async_to_lds_b128 (+asynccnt),
//       s_wait_dscnt, global_prefetch_b8.
// ---------------------------------------------------------------------------

#define SEQ     2048
#define HIDDEN  2048
#define NHEADS  16
#define HDIM    128

typedef __attribute__((ext_vector_type(16))) __bf16 v16bf;
typedef __attribute__((ext_vector_type(8)))  float  v8f;

__device__ __forceinline__ __bf16 to_bf16(float x) { return (__bf16)x; }

// Low 32 bits of a generic (flat) shared-memory address == LDS byte offset
// (ISA: aperture check uses addr[63:32]; LDS offset is the low 32 bits).
__device__ __forceinline__ unsigned lds_off(const void* p) {
  return (unsigned)(size_t)p;
}

// Async byte-copy global -> LDS, 16B per lane, payload at (base + goff).
__device__ __forceinline__ void async_b128(unsigned loff, unsigned goff,
                                           const void* base) {
  asm volatile("global_load_async_to_lds_b128 %0, %1, %2"
               :: "v"(loff), "v"(goff), "s"(base) : "memory");
}
__device__ __forceinline__ void async_b128_o(unsigned loff, unsigned goff,
                                             const void* base, int imm) {
  if (imm == 16)
    asm volatile("global_load_async_to_lds_b128 %0, %1, %2 offset:16"
                 :: "v"(loff), "v"(goff), "s"(base) : "memory");
  else if (imm == 32)
    asm volatile("global_load_async_to_lds_b128 %0, %1, %2 offset:32"
                 :: "v"(loff), "v"(goff), "s"(base) : "memory");
  else
    asm volatile("global_load_async_to_lds_b128 %0, %1, %2 offset:48"
                 :: "v"(loff), "v"(goff), "s"(base) : "memory");
}
__device__ __forceinline__ void wait_async0() {
  asm volatile("s_wait_asynccnt 0" ::: "memory");
}

union Frag16 { uint4 u[2]; v16bf v; };

// A-fragment (16x32, 16-bit): lane holds row = lane&15.
// elements 0..7  <-> K = 8*half + i ; elements 8..15 <-> K = 16 + 8*half + i
__device__ __forceinline__ v16bf frag_a_rowmajor(const __bf16* rowptr, int half) {
  Frag16 f;
  f.u[0] = *(const uint4*)(rowptr + 8 * half);
  f.u[1] = *(const uint4*)(rowptr + 16 + 8 * half);
  return f.v;
}

// B-fragment (32x16, 16-bit): lane holds col = lane&15.
// elements 0..7  <-> K = 8*half + i ; elements 8..15 <-> K = 16 + 8*half + i
__device__ __forceinline__ v16bf frag_b_kcontig(const __bf16* colptr, int half) {
  Frag16 f;
  f.u[0] = *(const uint4*)(colptr + 8 * half);
  f.u[1] = *(const uint4*)(colptr + 16 + 8 * half);
  return f.v;
}

// ---------------------------------------------------------------------------
// fp32 -> bf16 conversion (hidden_states)
// ---------------------------------------------------------------------------
__global__ __launch_bounds__(256) void cvt_f32_bf16(const float* __restrict__ in,
                                                    __bf16* __restrict__ out, int n) {
  int i = (blockIdx.x * 256 + threadIdx.x) * 4;
  if (i < n) {
    float4 v = *(const float4*)(in + i);
    out[i + 0] = to_bf16(v.x);
    out[i + 1] = to_bf16(v.y);
    out[i + 2] = to_bf16(v.z);
    out[i + 3] = to_bf16(v.w);
  }
}

// ---------------------------------------------------------------------------
// Tiled WMMA GEMM: C[M,N] = A(bf16)[M,K] @ B(f32)[K,N]
//   Block tile 128x128, 256 threads = 8 waves, K-step 32.
//   Each wave: 2 row-strips x 4 col-tiles = 8 WMMAs / K-step (B frags reused).
//   A tile staged with global_load_async_to_lds_b128.
//   EPI: 0 = store bf16, 1 = fused RoPE then bf16 (N tile == one head), 2 = f32
// ---------------------------------------------------------------------------
#define A_STRIDE 40   // bf16, padded (80B rows, 16B aligned)
#define B_STRIDE 40   // bf16, transposed [n][k], padded

template <int EPI>
__global__ __launch_bounds__(256)
void gemm_bf16_wmma(const __bf16* __restrict__ A, const float* __restrict__ B,
                    void* __restrict__ Cout, int M, int N, int K) {
  __shared__ __align__(16) char smem[34816];
  __bf16* a_lds = (__bf16*)smem;            // [128][A_STRIDE] = 10240 B
  __bf16* b_lds = (__bf16*)(smem + 10240);  // [128][B_STRIDE] = 10240 B

  const int tid   = threadIdx.x;
  const int lane  = tid & 31;
  const int wave  = tid >> 5;
  const int half  = lane >> 4;
  const int l15   = lane & 15;
  const int mbase = blockIdx.y * 128;
  const int nbase = blockIdx.x * 128;
  const int strip = wave & 3;   // rows strip*16 and strip*16+64
  const int ch    = wave >> 2;  // 64-col half

  const int a_row = tid >> 1, a_kc = (tid & 1) * 16;  // A: 128 rows x 32 k
  const int b_kr  = tid >> 3, b_nc = (tid & 7) * 16;  // B: 32 k x 128 n
  const unsigned a_loff = lds_off(a_lds + a_row * A_STRIDE + a_kc);

  v8f acc0[4] = {}, acc1[4] = {};

  for (int k0 = 0; k0 < K; k0 += 32) {
    // ---- A tile: async byte-copy (bf16, row-major, K contiguous) ----
    {
      unsigned goff = (unsigned)(((size_t)(mbase + a_row) * K + k0 + a_kc) * 2);
      async_b128(a_loff, goff, A);
      async_b128_o(a_loff, goff, A, 16);
    }
    // ---- B tile: f32 -> bf16, transposed to (n,k) ----
    float tmp[16];
    {
      const float* bp = B + (size_t)(k0 + b_kr) * N + nbase + b_nc;
      *(float4*)(tmp + 0)  = *(const float4*)(bp + 0);
      *(float4*)(tmp + 4)  = *(const float4*)(bp + 4);
      *(float4*)(tmp + 8)  = *(const float4*)(bp + 8);
      *(float4*)(tmp + 12) = *(const float4*)(bp + 12);
    }
    if (k0 + 32 < K)  // global_prefetch_b8 of next B K-step
      __builtin_prefetch(B + (size_t)(k0 + 32 + b_kr) * N + nbase + b_nc, 0, 0);
#pragma unroll
    for (int j = 0; j < 16; ++j)
      b_lds[(b_nc + j) * B_STRIDE + b_kr] = to_bf16(tmp[j]);
    wait_async0();
    __syncthreads();

    // ---- compute: 2 A-frags, 4 B-frags, 8 WMMAs ----
    const __bf16* arow0 = a_lds + (strip * 16 + l15) * A_STRIDE;
    v16bf af0 = frag_a_rowmajor(arow0, half);
    v16bf af1 = frag_a_rowmajor(arow0 + 64 * A_STRIDE, half);
#pragma unroll
    for (int ct = 0; ct < 4; ++ct) {
      const __bf16* bcol = b_lds + (ch * 64 + ct * 16 + l15) * B_STRIDE;
      v16bf bf = frag_b_kcontig(bcol, half);
      acc0[ct] = __builtin_amdgcn_wmma_f32_16x16x32_bf16(
          false, af0, false, bf, (short)0, acc0[ct], false, false);
      acc1[ct] = __builtin_amdgcn_wmma_f32_16x16x32_bf16(
          false, af1, false, bf, (short)0, acc1[ct], false, false);
    }
    __syncthreads();
  }

  if (EPI == 1) {
    // Fused RoPE epilogue: two 64x128 passes through LDS (one head wide).
    float* ctile = (float*)smem;  // [64][132] = 33792 B
    __bf16* out = (__bf16*)Cout;
    const float LOG_BASE = 9.210340371976184f;  // ln(10000)
#pragma unroll
    for (int p = 0; p < 2; ++p) {
      __syncthreads();
#pragma unroll
      for (int ct = 0; ct < 4; ++ct)
#pragma unroll
        for (int r = 0; r < 8; ++r) {
          float v = p ? acc1[ct][r] : acc0[ct][r];
          ctile[(strip * 16 + r + 8 * half) * 132 + ch * 64 + ct * 16 + l15] = v;
        }
      __syncthreads();
      for (int idx = tid; idx < 64 * 128; idx += 256) {
        int row = idx >> 7, d = idx & 127;
        float x  = ctile[row * 132 + d];
        float xo = ctile[row * 132 + (d ^ 64)];
        int grow = mbase + p * 64 + row;
        float invf = __expf(-(float)(d & 63) * (LOG_BASE / 64.0f));
        float sn, cs;
        __sincosf((float)grow * invf, &sn, &cs);
        float rot = (d < 64) ? -xo : xo;
        out[(size_t)grow * N + nbase + d] = to_bf16(x * cs + rot * sn);
      }
    }
  } else {
#pragma unroll
    for (int ct = 0; ct < 4; ++ct)
#pragma unroll
      for (int r = 0; r < 8; ++r) {
        size_t row = (size_t)(mbase + strip * 16 + r + 8 * half);
        size_t col = (size_t)(nbase + ch * 64 + ct * 16 + l15);
        if (EPI == 0) {
          ((__bf16*)Cout)[row * N + col]        = to_bf16(acc0[ct][r]);
          ((__bf16*)Cout)[(row + 64) * N + col] = to_bf16(acc1[ct][r]);
        } else {
          ((float*)Cout)[row * N + col]         = acc0[ct][r];
          ((float*)Cout)[(row + 64) * N + col]  = acc1[ct][r];
        }
      }
  }
}

// ---------------------------------------------------------------------------
// Flash attention per (head, 64-query block). 128 threads = 4 waves,
// each wave owns 16 query rows. 32 keys per iteration, causal.
// K tile staged with global_load_async_to_lds_b128.
// ---------------------------------------------------------------------------
#define K_STRIDE 136  // lds_k [32 keys][136]  (d contiguous)
#define V_STRIDE 40   // lds_vt [128 cols][40] (key contiguous)
#define P_STRIDE 40   // lds_p per-wave [16 rows][40]

__global__ __launch_bounds__(128)
void mla_attention(const __bf16* __restrict__ Q, const __bf16* __restrict__ Km,
                   const __bf16* __restrict__ V, __bf16* __restrict__ O) {
  __shared__ __align__(16) char smem[24064];
  __bf16* lds_k  = (__bf16*)smem;                   // 32*136*2 = 8704 B
  __bf16* lds_vt = (__bf16*)(smem + 8704);          // 128*40*2 = 10240 B
  __bf16* lds_p  = (__bf16*)(smem + 8704 + 10240);  // 4*16*40*2 = 5120 B

  const int tid  = threadIdx.x;
  const int lane = tid & 31;
  const int wave = tid >> 5;
  const int half = lane >> 4;
  const int l15  = lane & 15;
  const int h    = blockIdx.y;
  const int qb   = blockIdx.x;
  const int qbase = qb * 64 + wave * 16;
  __bf16* myp = lds_p + wave * 16 * P_STRIDE;

  // Per-thread K-tile copy coords (32 keys x 128 d, 64B chunk per thread)
  const int kt_r = tid >> 2, kt_d = (tid & 3) * 32;
  const unsigned k_loff = lds_off(lds_k + kt_r * K_STRIDE + kt_d);

  // Q fragments for this wave's 16 rows (direct from global, K-contiguous)
  v16bf qf[4];
  const __bf16* qrow = Q + (size_t)(qbase + l15) * HIDDEN + h * HDIM;
#pragma unroll
  for (int c = 0; c < 4; ++c) qf[c] = frag_a_rowmajor(qrow + c * 32, half);

  v8f o_acc[8] = {};
  float m_run[8], l_run[8];
#pragma unroll
  for (int r = 0; r < 8; ++r) { m_run[r] = -3.0e38f; l_run[r] = 0.0f; }

  const float scale = 0.08838834764831845f;  // 1/sqrt(128)
  const int nkb = qb * 2 + 2;                // causal: key blocks <= query block

  for (int kb = 0; kb < nkb; ++kb) {
    // ---- K tile via async copy: 4 x b128 per thread ----
    {
      unsigned goff =
          (unsigned)(((size_t)(kb * 32 + kt_r) * HIDDEN + h * HDIM + kt_d) * 2);
      async_b128(k_loff, goff, Km);
      async_b128_o(k_loff, goff, Km, 16);
      async_b128_o(k_loff, goff, Km, 32);
      async_b128_o(k_loff, goff, Km, 48);
    }
    // ---- V tile, transposed to [col][key] (manual: layout transform) ----
    {
      int key = tid & 31, cg = tid >> 5;
      const __bf16* src = V + (size_t)(kb * 32 + key) * HIDDEN + h * HDIM + cg * 32;
#pragma unroll
      for (int j = 0; j < 32; ++j) lds_vt[(cg * 32 + j) * V_STRIDE + key] = src[j];
    }
    wait_async0();
    __syncthreads();

    // ---- scores: 16 rows x 32 keys (2 C-frags, 8 WMMAs) ----
    v8f sc[2] = {};
#pragma unroll
    for (int kt = 0; kt < 2; ++kt) {
      const __bf16* krow = lds_k + (kt * 16 + l15) * K_STRIDE;
#pragma unroll
      for (int c = 0; c < 4; ++c) {
        v16bf bf = frag_b_kcontig(krow + c * 32, half);
        sc[kt] = __builtin_amdgcn_wmma_f32_16x16x32_bf16(
            false, qf[c], false, bf, (short)0, sc[kt], false, false);
      }
    }
    // ---- scale + causal mask ----
#pragma unroll
    for (int kt = 0; kt < 2; ++kt) {
      int key = kb * 32 + kt * 16 + l15;
#pragma unroll
      for (int r = 0; r < 8; ++r) {
        int row = qbase + r + 8 * half;
        float v = sc[kt][r] * scale;
        sc[kt][r] = (key > row) ? -1.0e9f : v;
      }
    }
    // ---- online softmax (row reductions across 16-lane halves) ----
    float mcur[8], lcur[8];
#pragma unroll
    for (int r = 0; r < 8; ++r) mcur[r] = fmaxf(sc[0][r], sc[1][r]);
#pragma unroll
    for (int msk = 1; msk <= 8; msk <<= 1)
#pragma unroll
      for (int r = 0; r < 8; ++r) mcur[r] = fmaxf(mcur[r], __shfl_xor(mcur[r], msk, 32));
    float fac[8];
#pragma unroll
    for (int r = 0; r < 8; ++r) {
      float mnew = fmaxf(m_run[r], mcur[r]);
      fac[r] = __expf(m_run[r] - mnew);
      m_run[r] = mnew;
      float p0 = __expf(sc[0][r] - mnew);
      float p1 = __expf(sc[1][r] - mnew);
      sc[0][r] = p0; sc[1][r] = p1;
      lcur[r] = p0 + p1;
    }
#pragma unroll
    for (int msk = 1; msk <= 8; msk <<= 1)
#pragma unroll
      for (int r = 0; r < 8; ++r) lcur[r] += __shfl_xor(lcur[r], msk, 32);
#pragma unroll
    for (int r = 0; r < 8; ++r) l_run[r] = l_run[r] * fac[r] + lcur[r];
#pragma unroll
    for (int f = 0; f < 8; ++f)
#pragma unroll
      for (int r = 0; r < 8; ++r) o_acc[f][r] *= fac[r];

    // ---- stage P (C-layout -> A-layout transpose via per-wave LDS) ----
#pragma unroll
    for (int kt = 0; kt < 2; ++kt)
#pragma unroll
      for (int r = 0; r < 8; ++r)
        myp[(r + 8 * half) * P_STRIDE + kt * 16 + l15] = to_bf16(sc[kt][r]);
    asm volatile("s_wait_dscnt 0" ::: "memory");

    // ---- O += P @ V : 8 WMMAs ----
    v16bf pf = frag_a_rowmajor(myp + l15 * P_STRIDE, half);
#pragma unroll
    for (int f = 0; f < 8; ++f) {
      const __bf16* vcol = lds_vt + (f * 16 + l15) * V_STRIDE;
      v16bf vf = frag_b_kcontig(vcol, half);
      o_acc[f] = __builtin_amdgcn_wmma_f32_16x16x32_bf16(
          false, pf, false, vf, (short)0, o_acc[f], false, false);
    }
    __syncthreads();
  }

  // ---- normalize + store attn output (bf16) ----
#pragma unroll
  for (int f = 0; f < 8; ++f)
#pragma unroll
    for (int r = 0; r < 8; ++r) {
      float v = o_acc[f][r] / l_run[r];
      O[(size_t)(qbase + r + 8 * half) * HIDDEN + h * HDIM + f * 16 + l15] = to_bf16(v);
    }
}

// ---------------------------------------------------------------------------
// Launcher
// ---------------------------------------------------------------------------
extern "C" void kernel_launch(void* const* d_in, const int* in_sizes, int n_in,
                              void* d_out, int out_size, void* d_ws, size_t ws_size,
                              hipStream_t stream) {
  const float* hs = (const float*)d_in[0];
  const float* wq = (const float*)d_in[1];
  const float* wk = (const float*)d_in[2];
  const float* wv = (const float*)d_in[3];
  const float* wo = (const float*)d_in[4];
  // d_in[5] = attention_mask: pure causal, applied analytically in-kernel.

  char* ws = (char*)d_ws;
  __bf16* hs_bf = (__bf16*)(ws);                        // 8 MB
  __bf16* Qb    = (__bf16*)(ws + ((size_t)8  << 20));   // 8 MB (RoPE'd)
  __bf16* Kb    = (__bf16*)(ws + ((size_t)16 << 20));   // 8 MB (RoPE'd)
  __bf16* Vb    = (__bf16*)(ws + ((size_t)24 << 20));   // 8 MB
  __bf16* Ab    = (__bf16*)(ws + ((size_t)32 << 20));   // 8 MB attn out

  const int n = SEQ * HIDDEN;
  cvt_f32_bf16<<<n / (256 * 4), 256, 0, stream>>>(hs, hs_bf, n);

  dim3 g(HIDDEN / 128, SEQ / 128), b(256);
  gemm_bf16_wmma<1><<<g, b, 0, stream>>>(hs_bf, wq, Qb, SEQ, HIDDEN, HIDDEN);
  gemm_bf16_wmma<1><<<g, b, 0, stream>>>(hs_bf, wk, Kb, SEQ, HIDDEN, HIDDEN);
  gemm_bf16_wmma<0><<<g, b, 0, stream>>>(hs_bf, wv, Vb, SEQ, HIDDEN, HIDDEN);

  dim3 ga(SEQ / 64, NHEADS);
  mla_attention<<<ga, dim3(128), 0, stream>>>(Qb, Kb, Vb, Ab);

  gemm_bf16_wmma<2><<<g, b, 0, stream>>>(Ab, wo, (float*)d_out, SEQ, HIDDEN, HIDDEN);
}
